// SAGENN_55783035240979
// MI455X (gfx1250) — matrix operand order repaired and checked by reading, not compiled, not run on hardware
//
#include <hip/hip_runtime.h>

typedef __attribute__((ext_vector_type(16))) _Float16 v16h;
typedef __attribute__((ext_vector_type(8)))  float    v8f;

#define N_NODES  100000
#define N_EDGES  1600000
#define N_GRAPHS 128
#define FDIM     64
#define N_TILES  (N_NODES / 16)
#define EPS      1e-5f
#define NEG      0.01f

// ---------------------------------------------------------------- fill
__global__ void k_fill(float* p, float v, long n) {
    long i = (long)blockIdx.x * blockDim.x + threadIdx.x;
    if (i < n) p[i] = v;
}

// ---------------------------------------------------------------- edge gather/scatter
// 16 lanes per edge, 4 features per lane (float4). BN of previous layer folded in
// via per-feature scale/shift (identity for layer 0).
__global__ void __launch_bounds__(256) k_gather(
    const float* __restrict__ xin, const float* __restrict__ scale,
    const float* __restrict__ shift, const int* __restrict__ ei,
    float* __restrict__ agg, float* __restrict__ degp, int doDeg)
{
    long t = (long)blockIdx.x * blockDim.x + threadIdx.x;
    long e = t >> 4;
    if (e >= N_EDGES) return;
    int fq = ((int)t & 15) * 4;
    int src = ei[e];
    int dst = ei[N_EDGES + e];
    const float4 xv = *(const float4*)(xin + (size_t)src * FDIM + fq);
    const float4 s4 = *(const float4*)(scale + fq);
    const float4 h4 = *(const float4*)(shift + fq);
    float* ap = agg + (size_t)dst * FDIM + fq;
    atomicAdd(ap + 0, xv.x * s4.x + h4.x);
    atomicAdd(ap + 1, xv.y * s4.y + h4.y);
    atomicAdd(ap + 2, xv.z * s4.z + h4.z);
    atomicAdd(ap + 3, xv.w * s4.w + h4.w);
    if (doDeg && ((t & 15) == 0)) atomicAdd(degp + dst, 1.0f);
}

// ---------------------------------------------------------------- SAGE transform (WMMA)
// h = leaky( (agg/deg) @ Wl + bl + BN(prev) @ Wr ), plus per-feature sum / sumsq.
// One wave per 16-node tile; v_wmma_f32_16x16x32_f16, 2 K-blocks x 4 N-blocks x 2 mats.
__global__ void __launch_bounds__(256) k_transform(
    const float* __restrict__ prev, const float* __restrict__ scale,
    const float* __restrict__ shift, const float* __restrict__ agg,
    const float* __restrict__ degp, const float* __restrict__ Wl,
    const float* __restrict__ bl, const float* __restrict__ Wr,
    float* __restrict__ hout, float* __restrict__ sumOut, float* __restrict__ sqOut)
{
    __shared__ _Float16 sWl[FDIM * FDIM];
    __shared__ _Float16 sWr[FDIM * FDIM];
    __shared__ float sScale[FDIM], sShift[FDIM], sBias[FDIM];

    int tid = threadIdx.x;
    for (int i = tid; i < FDIM * FDIM; i += 256) {
        sWl[i] = (_Float16)Wl[i];
        sWr[i] = (_Float16)Wr[i];
    }
    if (tid < FDIM) { sScale[tid] = scale[tid]; sShift[tid] = shift[tid]; sBias[tid] = bl[tid]; }
    __syncthreads();

    const int lane = tid & 31;          // wave32
    const int r    = lane & 15;         // row (A/M) or column (B/N, C/N) index
    const int hi   = lane >> 4;         // half-wave selector
    int gwave = ((int)blockIdx.x * 256 + tid) >> 5;
    int nwave = ((int)gridDim.x * 256) >> 5;

    float sAcc[4] = {0.f, 0.f, 0.f, 0.f};
    float qAcc[4] = {0.f, 0.f, 0.f, 0.f};

    for (int tile = gwave; tile < N_TILES; tile += nwave) {
        int row = tile * 16 + r;
        float dinv = 1.0f / fmaxf(degp[row], 1.0f);
        const float* arow = agg  + (size_t)row * FDIM;
        const float* prow = prev + (size_t)row * FDIM;

        // A fragments: 16x32 f16 layout (ISA 7.12.2): lane m holds row M=m;
        // hi=0 -> K = {b..b+7, b+16..b+23}, hi=1 -> +8, b = 32*kb.
        v16h aL[2], aR[2];
#pragma unroll
        for (int kb = 0; kb < 2; ++kb) {
            int base = kb * 32 + hi * 8;
#pragma unroll
            for (int j = 0; j < 8; ++j) {
                int k0 = base + j, k1 = base + 16 + j;
                aL[kb][j]     = (_Float16)(arow[k0] * dinv);
                aL[kb][8 + j] = (_Float16)(arow[k1] * dinv);
                aR[kb][j]     = (_Float16)(prow[k0] * sScale[k0] + sShift[k0]);
                aR[kb][8 + j] = (_Float16)(prow[k1] * sScale[k1] + sShift[k1]);
            }
        }

#pragma unroll
        for (int nb = 0; nb < 4; ++nb) {
            int n = nb * 16 + r;
            // B fragments: 32x16 f16 layout: lane holds column N=lane%16, K = 8*hi + (j&7) + 16*(j>>3).
            v16h bLf[2], bRf[2];
#pragma unroll
            for (int kb = 0; kb < 2; ++kb) {
#pragma unroll
                for (int j = 0; j < 16; ++j) {
                    int k = kb * 32 + hi * 8 + (j & 7) + (j >> 3) * 16;
                    bLf[kb][j] = sWl[k * FDIM + n];
                    bRf[kb][j] = sWr[k * FDIM + n];
                }
            }
            v8f c;
            float bias = sBias[n];
#pragma unroll
            for (int i = 0; i < 8; ++i) c[i] = bias;
            c = __builtin_amdgcn_wmma_f32_16x16x32_f16(false, aL[0], false, bLf[0], (short)0, c, false, false);
            c = __builtin_amdgcn_wmma_f32_16x16x32_f16(false, aL[1], false, bLf[1], (short)0, c, false, false);
            c = __builtin_amdgcn_wmma_f32_16x16x32_f16(false, aR[0], false, bRf[0], (short)0, c, false, false);
            c = __builtin_amdgcn_wmma_f32_16x16x32_f16(false, aR[1], false, bRf[1], (short)0, c, false, false);

            // C layout: VGPR i -> M = i + 8*hi, N = lane%16.
#pragma unroll
            for (int i = 0; i < 8; ++i) {
                float v = c[i];
                v = (v >= 0.f) ? v : NEG * v;
                int orow = tile * 16 + hi * 8 + i;
                hout[(size_t)orow * FDIM + n] = v;
                sAcc[nb] += v;
                qAcc[nb] += v * v;
            }
        }
    }
#pragma unroll
    for (int nb = 0; nb < 4; ++nb) {
        atomicAdd(sumOut + nb * 16 + r, sAcc[nb]);
        atomicAdd(sqOut  + nb * 16 + r, qAcc[nb]);
    }
}

// ---------------------------------------------------------------- BN stats -> scale/shift
__global__ void k_bnfin(const float* __restrict__ sum, const float* __restrict__ sq,
                        const float* __restrict__ gamma, const float* __restrict__ beta,
                        float* __restrict__ scale, float* __restrict__ shift)
{
    int f = threadIdx.x;
    if (f < FDIM) {
        float mu  = sum[f] * (1.0f / N_NODES);
        float var = sq[f]  * (1.0f / N_NODES) - mu * mu;
        float s = gamma[f] * rsqrtf(var + EPS);
        scale[f] = s;
        shift[f] = beta[f] - mu * s;
    }
}

// ---------------------------------------------------------------- graph mean-pool (LDS staging)
__global__ void __launch_bounds__(256) k_pool(
    const float* __restrict__ h, const float* __restrict__ scale,
    const float* __restrict__ shift, const int* __restrict__ batch,
    float* __restrict__ pooled, float* __restrict__ cnt)
{
    __shared__ float sp[N_GRAPHS * FDIM];   // 32 KB
    __shared__ float sc[N_GRAPHS];
    int tid = threadIdx.x;
    for (int i = tid; i < N_GRAPHS * FDIM; i += 256) sp[i] = 0.f;
    for (int i = tid; i < N_GRAPHS; i += 256) sc[i] = 0.f;
    __syncthreads();

    const int CH = 1024;
    int n0 = (int)blockIdx.x * CH;
    int nend = n0 + CH; if (nend > N_NODES) nend = N_NODES;
    int fq = (tid & 15) * 4;
    const float4 s4 = *(const float4*)(scale + fq);
    const float4 h4 = *(const float4*)(shift + fq);
    for (int i = n0 + (tid >> 4); i < nend; i += 16) {
        int g = batch[i];
        const float4 v = *(const float4*)(h + (size_t)i * FDIM + fq);
        float* p = sp + g * FDIM + fq;
        atomicAdd(p + 0, v.x * s4.x + h4.x);
        atomicAdd(p + 1, v.y * s4.y + h4.y);
        atomicAdd(p + 2, v.z * s4.z + h4.z);
        atomicAdd(p + 3, v.w * s4.w + h4.w);
        if ((tid & 15) == 0) atomicAdd(sc + g, 1.0f);
    }
    __syncthreads();
    for (int i = tid; i < N_GRAPHS * FDIM; i += 256) atomicAdd(pooled + i, sp[i]);
    for (int i = tid; i < N_GRAPHS; i += 256) atomicAdd(cnt + i, sc[i]);
}

// ---------------------------------------------------------------- concat(pooled/cnt, gfeat)
__global__ void k_buildz(const float* __restrict__ pooled, const float* __restrict__ cnt,
                         const float* __restrict__ gf, float* __restrict__ z)
{
    int i = (int)blockIdx.x * blockDim.x + threadIdx.x;
    if (i >= N_GRAPHS * 128) return;
    int g = i >> 7, j = i & 127;
    float v = (j < 64) ? pooled[g * 64 + j] / fmaxf(cnt[g], 1.0f)
                       : gf[g * 64 + (j - 64)];
    z[i] = v;
}

// ---------------------------------------------------------------- tiny dense layer
__global__ void k_dense(const float* __restrict__ in, const float* __restrict__ W,
                        const float* __restrict__ b, float* __restrict__ out,
                        int Fin, int Fout, int act)
{
    int i = (int)blockIdx.x * blockDim.x + threadIdx.x;
    if (i >= N_GRAPHS * Fout) return;
    int g = i / Fout, n = i % Fout;
    float acc = b[n];
    const float* row = in + (size_t)g * Fin;
    for (int k = 0; k < Fin; ++k) acc += row[k] * W[k * Fout + n];
    if (act) acc = (acc >= 0.f) ? acc : NEG * acc;
    out[i] = acc;
}

// ---------------------------------------------------------------- driver
extern "C" void kernel_launch(void* const* d_in, const int* in_sizes, int n_in,
                              void* d_out, int out_size, void* d_ws, size_t ws_size,
                              hipStream_t stream)
{
    const float* x   = (const float*)d_in[0];
    const int*   ei  = (const int*)  d_in[1];
    const float* gf  = (const float*)d_in[2];
    const int*   bat = (const int*)  d_in[3];
    const float *Wl0 = (const float*)d_in[4],  *bl0 = (const float*)d_in[5],
                *Wr0 = (const float*)d_in[6],  *g0  = (const float*)d_in[7],
                *b0  = (const float*)d_in[8];
    const float *Wl1 = (const float*)d_in[9],  *bl1 = (const float*)d_in[10],
                *Wr1 = (const float*)d_in[11], *g1  = (const float*)d_in[12],
                *b1  = (const float*)d_in[13];
    const float *Wm0 = (const float*)d_in[14], *bm0 = (const float*)d_in[15],
                *Wm1 = (const float*)d_in[16], *bm1 = (const float*)d_in[17],
                *Wm2 = (const float*)d_in[18], *bm2 = (const float*)d_in[19];

    float* ws   = (float*)d_ws;
    float* agg  = ws;                   // 6,400,000
    float* h0   = ws + 6400000;         // 6,400,000
    float* h1   = ws + 12800000;        // 6,400,000
    float* dg   = ws + 19200000;        // 100,000
    float* sum0 = ws + 19300000;        // stats block: sum0,sq0,sum1,sq1 (4x64)
    float* sq0  = sum0 + 64;
    float* sum1 = sq0 + 64;
    float* sq1  = sum1 + 64;
    float* scId = sq1 + 64;  float* shId = scId + 64;
    float* sc0  = shId + 64; float* sh0  = sc0 + 64;
    float* sc1  = sh0 + 64;  float* sh1  = sc1 + 64;
    float* pooled = sh1 + 64;           // 8192
    float* cnt  = pooled + 8192;        // 128
    float* z    = cnt + 128;            // 16384
    float* z1   = z + 16384;            // 32768
    float* z2   = z1 + 32768;           // 16384

    auto fill = [&](float* p, float v, long n) {
        k_fill<<<(unsigned)((n + 255) / 256), 256, 0, stream>>>(p, v, n);
    };
    fill(agg, 0.f, 6400000);
    fill(dg, 0.f, 100000);
    fill(sum0, 0.f, 256);               // sum0,sq0,sum1,sq1
    fill(scId, 1.f, 64);
    fill(shId, 0.f, 64);
    fill(pooled, 0.f, 8192 + 128);      // pooled + cnt

    const unsigned gatherBlocks = (unsigned)(((long)N_EDGES * 16) / 256);  // 100000

    // ---- layer 0 ----
    k_gather<<<gatherBlocks, 256, 0, stream>>>(x, scId, shId, ei, agg, dg, 1);
    k_transform<<<782, 256, 0, stream>>>(x, scId, shId, agg, dg, Wl0, bl0, Wr0, h0, sum0, sq0);
    k_bnfin<<<1, 64, 0, stream>>>(sum0, sq0, g0, b0, sc0, sh0);

    // ---- layer 1 (BN of layer 0 folded into gather/transform reads) ----
    fill(agg, 0.f, 6400000);
    k_gather<<<gatherBlocks, 256, 0, stream>>>(h0, sc0, sh0, ei, agg, dg, 0);
    k_transform<<<782, 256, 0, stream>>>(h0, sc0, sh0, agg, dg, Wl1, bl1, Wr1, h1, sum1, sq1);
    k_bnfin<<<1, 64, 0, stream>>>(sum1, sq1, g1, b1, sc1, sh1);

    // ---- pool + MLP ----
    k_pool<<<(N_NODES + 1023) / 1024, 256, 0, stream>>>(h1, sc1, sh1, bat, pooled, cnt);
    k_buildz<<<(N_GRAPHS * 128 + 255) / 256, 256, 0, stream>>>(pooled, cnt, gf, z);
    k_dense<<<(N_GRAPHS * 256 + 255) / 256, 256, 0, stream>>>(z,  Wm0, bm0, z1, 128, 256, 1);
    k_dense<<<(N_GRAPHS * 128 + 255) / 256, 256, 0, stream>>>(z1, Wm1, bm1, z2, 256, 128, 1);
    k_dense<<<1, 256, 0, stream>>>(z2, Wm2, bm2, (float*)d_out, 128, 2, 0);
}
